// MPNN_10428180594979
// MI455X (gfx1250) — matrix operand order repaired and checked by reading, not compiled, not run on hardware
//
#include <hip/hip_runtime.h>
#include <hip/hip_bf16.h>

// ---------------- types / helpers ----------------
typedef __attribute__((ext_vector_type(16))) __bf16 bf16x16;
typedef __attribute__((ext_vector_type(8)))  float  v8f;
typedef __attribute__((ext_vector_type(2)))  int    i32x2;
typedef __attribute__((ext_vector_type(4)))  int    i32x4;

#define HID 128
#define SK  136   // padded K stride in halves (multiple of 8 -> 16B aligned rows)

#if __has_builtin(__builtin_amdgcn_global_load_async_to_lds_b64) && \
    __has_builtin(__builtin_amdgcn_global_load_async_to_lds_b128)
#define HAVE_ASYNC_LDS 1
#endif

typedef __attribute__((address_space(1))) void* gas_t;
typedef __attribute__((address_space(3))) void* las_t;
typedef __attribute__((address_space(1))) i32x2* g2_t;
typedef __attribute__((address_space(3))) i32x2* l2_t;
typedef __attribute__((address_space(1))) i32x4* g4_t;
typedef __attribute__((address_space(3))) i32x4* l4_t;

union FragB { bf16x16 v; uint4 q[2]; unsigned short h[16]; };

__device__ __forceinline__ float swishf(float x) { return x / (1.0f + __expf(-x)); }

__device__ __forceinline__ unsigned int f2bfu(float f) {
    union { float f; unsigned int u; } v; v.f = f;
    unsigned int u = v.u;
    return (u + 0x7FFFu + ((u >> 16) & 1u)) >> 16;   // RNE bf16
}

__device__ __forceinline__ v8f splat8(float x) {
    v8f v = {x, x, x, x, x, x, x, x};
    return v;
}

__device__ __forceinline__ v8f wmma_bf16(bf16x16 a, bf16x16 b, v8f c) {
    return __builtin_amdgcn_wmma_f32_16x16x32_bf16(false, a, false, b, (short)0, c,
                                                   false, false);
}

// copy 8B per lane, global -> LDS (async DMA when available)
__device__ __forceinline__ void cp8_g2l(const unsigned short* g, unsigned short* l) {
#ifdef HAVE_ASYNC_LDS
    __builtin_amdgcn_global_load_async_to_lds_b64((g2_t)(gas_t)g, (l2_t)(las_t)l, 0, 0);
#else
    *(uint2*)l = *(const uint2*)g;
#endif
}
// copy 16B per lane, global -> LDS
__device__ __forceinline__ void cp16_g2l(const unsigned short* g, unsigned short* l) {
#ifdef HAVE_ASYNC_LDS
    __builtin_amdgcn_global_load_async_to_lds_b128((g4_t)(gas_t)g, (l4_t)(las_t)l, 0, 0);
#else
    *(uint4*)l = *(const uint4*)g;
#endif
}
// wait for this wave's outstanding async LDS transfers
__device__ __forceinline__ void async_fence() {
#ifdef HAVE_ASYNC_LDS
#if __has_builtin(__builtin_amdgcn_s_wait_asynccnt)
    __builtin_amdgcn_s_wait_asynccnt(0);
    asm volatile("" ::: "memory");
#else
    asm volatile("s_wait_asynccnt 0x0" ::: "memory");
#endif
#endif
}

// A tile: row-major [m (=lane&15)][K], bf16, stride SK halves.
__device__ __forceinline__ bf16x16 frag_a(const unsigned short* tile, int lo, int hi, int kc) {
    FragB f;
    const unsigned short* p = tile + lo * SK + kc + (hi << 3);
    f.q[0] = *(const uint4*)(p);
    f.q[1] = *(const uint4*)(p + 16);
    return f.v;
}

// B tile: column-major [n][K] bf16, stride SK halves.
__device__ __forceinline__ bf16x16 frag_b(const unsigned short* tile, int lo, int hi, int kc, int nb) {
    FragB f;
    const unsigned short* p = tile + (nb + lo) * SK + kc + (hi << 3);
    f.q[0] = *(const uint4*)(p);
    f.q[1] = *(const uint4*)(p + 16);
    return f.v;
}

// cooperative copy of one 128x128 col-major bf16 weight matrix into padded LDS
__device__ __forceinline__ void load_w(unsigned short* dst, const unsigned short* src, int tid) {
    for (int i = tid; i < 128 * 16; i += 128) {
        const int n  = i >> 4;
        const int k8 = (i & 15) << 3;
        cp16_g2l(src + (n << 7) + k8, dst + n * SK + k8);
    }
}

// ---------------- small prep kernels ----------------
__global__ void prep_u_kernel(const float* __restrict__ inp, float* __restrict__ un) {
    const int k = blockIdx.x * 256 + threadIdx.x;
    if (k < 65536) un[k] = inp[(size_t)(k >> 14) * 49152 + (k & 16383)];
}

// convert per-layer weights to bf16, col-major [n][k]
// mats: 0=Wt 1=Ws 2=Wm2 3=Wuf 4=Wua 5=Wu2
__global__ void conv_w_kernel(const float* __restrict__ msg1W, const float* __restrict__ msg2W,
                              const float* __restrict__ upd1W, const float* __restrict__ upd2W,
                              unsigned short* __restrict__ wbf) {
    const int gid = blockIdx.x * 256 + threadIdx.x;
    if (gid >= 6 * 6 * 16384) return;
    const int l   = gid / (6 * 16384);
    const int rem = gid - l * 6 * 16384;
    const int mat = rem >> 14;
    const int e   = rem & 16383;
    const int n   = e >> 7;
    const int k   = e & 127;
    float v;
    switch (mat) {
        case 0:  v = msg1W[(size_t)l * 264 * 128 + k * 128 + n];          break;
        case 1:  v = msg1W[(size_t)l * 264 * 128 + (128 + k) * 128 + n];  break;
        case 2:  v = msg2W[(size_t)l * 128 * 128 + k * 128 + n];          break;
        case 3:  v = upd1W[(size_t)l * 261 * 128 + k * 128 + n];          break;
        case 4:  v = upd1W[(size_t)l * 261 * 128 + (128 + k) * 128 + n];  break;
        default: v = upd2W[(size_t)l * 128 * 128 + k * 128 + n];          break;
    }
    wbf[gid] = (unsigned short)f2bfu(v);
}

// ---------------- embedding MLP (8->128->128), one block per node ----------------
__global__ __launch_bounds__(128)
void embed_kernel(const float* __restrict__ un, const float* __restrict__ cp,
                  const float* __restrict__ eW1, const float* __restrict__ eb1,
                  const float* __restrict__ eW2, const float* __restrict__ eb2,
                  float* __restrict__ fA, unsigned short* __restrict__ fbf) {
    const int node = blockIdx.x;
    const int tid  = threadIdx.x;
    const int b = node >> 14, loc = node & 16383, r = loc >> 7, c = loc & 127;
    float e[8];
    e[0] = un[node];
    e[1] = (float)r * (1.0f / 127.0f) * cp[b * 5 + 1];
    e[2] = (float)c * (1.0f / 127.0f) * cp[b * 5 + 0];
#pragma unroll
    for (int p = 0; p < 5; ++p) e[3 + p] = cp[b * 5 + p];
    __shared__ float h[128];
    float a = eb1[tid];
#pragma unroll
    for (int k = 0; k < 8; ++k) a += e[k] * eW1[k * 128 + tid];
    h[tid] = swishf(a);
    __syncthreads();
    float a2 = eb2[tid];
    for (int k = 0; k < 128; ++k) a2 += h[k] * eW2[k * 128 + tid];
    const float r2 = swishf(a2);
    fA[(size_t)node * HID + tid]  = r2;
    fbf[(size_t)node * HID + tid] = (unsigned short)f2bfu(r2);
}

// ---------------- one MPNN layer: messages (8 dirs) + aggregate + update ----------------
__global__ __launch_bounds__(128)
void layer_kernel(const float* __restrict__ fA, float* __restrict__ fB,
                  const unsigned short* __restrict__ fbf,
                  const float* __restrict__ un,
                  const unsigned short* __restrict__ wl,
                  const float* __restrict__ msg1W, const float* __restrict__ msg1b,
                  const float* __restrict__ msg2b,
                  const float* __restrict__ upd1W, const float* __restrict__ upd1b,
                  const float* __restrict__ upd2b,
                  const float* __restrict__ cp) {
    __shared__ __attribute__((aligned(16))) unsigned short wb0[128 * SK];
    __shared__ __attribute__((aligned(16))) unsigned short wb1[128 * SK];
    __shared__ __attribute__((aligned(16))) unsigned short wb2[128 * SK];
    __shared__ __attribute__((aligned(16))) unsigned short ftile[4][16 * SK];
    __shared__ __attribute__((aligned(16))) unsigned short stile[4][2][16 * SK]; // double-buffered
    __shared__ float gw[8 * 128];
    __shared__ float pconst[128];   // direction-independent part of msg1 bias vector
    __shared__ float pvec[128];
    __shared__ float m2bs[128];
    __shared__ float duv[4][16];
    __shared__ float vmask[4][16];

    const int tid  = threadIdx.x;
    const int lane = tid & 31;
    const int wave = tid >> 5;
    const int lo   = lane & 15;
    const int hi   = lane >> 4;
    const int base = blockIdx.x * 64;          // 64 consecutive target nodes per WG
    const int b    = base >> 14;
    const int loc  = base & 16383;
    const int r    = loc >> 7;                 // grid row, uniform in WG
    const int myc0   = (loc & 127) + wave * 16;
    const int mybase = base + wave * 16;

    float cpv[5];
#pragma unroll
    for (int p = 0; p < 5; ++p) cpv[p] = cp[b * 5 + p];

    // stage phase-1 weights + geo weight rows + msg2 bias
    for (int i = tid; i < 1024; i += 128) gw[i] = msg1W[256 * 128 + i];
    m2bs[tid] = msg2b[tid];
    load_w(wb0, wl + 0 * 16384, tid);   // W_t
    load_w(wb1, wl + 1 * 16384, tid);   // W_s
    load_w(wb2, wl + 2 * 16384, tid);   // msg2_W
    {   // pconst[n] = msg1_b[n] + sum_p params[p]*Wg[3+p][n]   (gw rows written by same tid)
        const int n = tid;
        float g = msg1b[n];
#pragma unroll
        for (int p = 0; p < 5; ++p) g += cpv[p] * gw[(3 + p) * 128 + n];
        pconst[n] = g;
    }

    unsigned short* ft  = &ftile[wave][0];
    unsigned short* stA = &stile[wave][0][0];
    unsigned short* stB = &stile[wave][1][0];

    const int DX[8] = {-1, -1, -1, 0, 0, 1, 1, 1};
    const int DY[8] = {-1, 0, 1, -1, 1, -1, 0, 1};

    // async neighbor-tile fetch (clamped when invalid; masked at aggregation)
    auto issue_tile = [&](int dx, int dy, unsigned short* dst) {
        const bool dxok = ((unsigned)(r + dx)) < 128u;
        for (int m = 0; m < 16; ++m) {
            const int  cs = myc0 + m + dy;
            const bool ok = dxok && ((unsigned)cs < 128u);
            const int  sn = ok ? (mybase + m + dx * 128 + dy) : (mybase + m);
            cp8_g2l(fbf + (size_t)sn * HID + lane * 4, dst + m * SK + lane * 4);
        }
    };

    // per-wave f[tgt] tile (A layout, bf16) straight from the bf16 mirror
    for (int m = 0; m < 16; ++m)
        cp8_g2l(fbf + (size_t)(mybase + m) * HID + lane * 4, ft + m * SK + lane * 4);
    issue_tile(DX[0], DY[0], stA);      // prologue: direction-0 tile in flight
    async_fence();
    __syncthreads();

    // T = f_t @ W_t (hoisted out of the direction loop)
    v8f Treg[8];
#pragma unroll
    for (int t = 0; t < 8; ++t) {
        v8f c = splat8(0.0f);
#pragma unroll
        for (int kc = 0; kc < 128; kc += 32)
            c = wmma_bf16(frag_a(ft, lo, hi, kc), frag_b(wb0, lo, hi, kc, t * 16), c);
        Treg[t] = c;
    }

    v8f aggr[8];
#pragma unroll
    for (int t = 0; t < 8; ++t) aggr[t] = splat8(0.0f);

    // ---- direction loop: barrier-free, software-pipelined async gathers ----
    for (int d = 0; d < 8; ++d) {
        unsigned short* sc = (d & 1) ? stB : stA;
        async_fence();                         // current buffer resident
        if (d < 7) issue_tile(DX[d + 1], DY[d + 1], (d & 1) ? stA : stB);

        const int dx = DX[d], dy = DY[d];
        if (lane < 16) {
            const int  m  = lane;
            const int  cs = myc0 + m + dy;
            const bool ok = (((unsigned)(r + dx)) < 128u) && ((unsigned)cs < 128u);
            const int  sn = ok ? (mybase + m + dx * 128 + dy) : (mybase + m);
            duv[wave][m]   = un[mybase + m] - un[sn];
            vmask[wave][m] = ok ? 1.0f : 0.0f;
        }
        const float dpx = -(float)dx * cpv[1] * (1.0f / 127.0f);
        const float dpy = -(float)dy * cpv[0] * (1.0f / 127.0f);

        // msg layer 1: T + f_s@W_s + du*Wg0 + (pconst + dpx*Wg1 + dpy*Wg2)
        v8f h1c[8];
#pragma unroll
        for (int t = 0; t < 8; ++t) {
            const int   n   = t * 16 + lo;
            const float wg0 = gw[n];
            const float gv  = pconst[n] + dpx * gw[128 + n] + dpy * gw[256 + n];
            v8f c = Treg[t];
#pragma unroll
            for (int i = 0; i < 8; ++i) c[i] += duv[wave][i + hi * 8] * wg0 + gv;
#pragma unroll
            for (int kc = 0; kc < 128; kc += 32)
                c = wmma_bf16(frag_a(sc, lo, hi, kc), frag_b(wb1, lo, hi, kc, t * 16), c);
            h1c[t] = c;
        }
        // swish(h1) -> A layout in sc (fS fully consumed above)
#pragma unroll
        for (int t = 0; t < 8; ++t) {
            const int n = t * 16 + lo;
#pragma unroll
            for (int i = 0; i < 8; ++i)
                sc[(i + hi * 8) * SK + n] = (unsigned short)f2bfu(swishf(h1c[t][i]));
        }
        // msg layer 2 + masked aggregation
#pragma unroll
        for (int t = 0; t < 8; ++t) {
            const int n = t * 16 + lo;
            v8f c = splat8(m2bs[n]);
#pragma unroll
            for (int kc = 0; kc < 128; kc += 32)
                c = wmma_bf16(frag_a(sc, lo, hi, kc), frag_b(wb2, lo, hi, kc, t * 16), c);
#pragma unroll
            for (int i = 0; i < 8; ++i)
                aggr[t][i] += vmask[wave][i + hi * 8] * swishf(c[i]);
        }
    }

    // ---------------- phase 2: node update ----------------
    __syncthreads();
    load_w(wb0, wl + 3 * 16384, tid);   // Wu_f
    load_w(wb1, wl + 4 * 16384, tid);   // Wu_a
    load_w(wb2, wl + 5 * 16384, tid);   // upd2_W
    {
        const int n = tid;
        float g = upd1b[n];
#pragma unroll
        for (int p = 0; p < 5; ++p) g += cpv[p] * upd1W[(256 + p) * 128 + n];
        pvec[n] = g;
    }
    // agg / deg -> A layout in stA
    const float degr = 3.0f - (float)((r == 0) + (r == 127));
#pragma unroll
    for (int t = 0; t < 8; ++t) {
        const int n = t * 16 + lo;
#pragma unroll
        for (int i = 0; i < 8; ++i) {
            const int m  = i + hi * 8;
            const int cc = myc0 + m;
            const float deg = degr * (3.0f - (float)((cc == 0) + (cc == 127))) - 1.0f;
            stA[m * SK + n] = (unsigned short)f2bfu(aggr[t][i] / deg);
        }
    }
    async_fence();
    __syncthreads();

    v8f u1c[8];
#pragma unroll
    for (int t = 0; t < 8; ++t) {
        const int n = t * 16 + lo;
        v8f c = splat8(pvec[n]);
#pragma unroll
        for (int kc = 0; kc < 128; kc += 32)
            c = wmma_bf16(frag_a(ft, lo, hi, kc), frag_b(wb0, lo, hi, kc, t * 16), c);
#pragma unroll
        for (int kc = 0; kc < 128; kc += 32)
            c = wmma_bf16(frag_a(stA, lo, hi, kc), frag_b(wb1, lo, hi, kc, t * 16), c);
        u1c[t] = c;
    }
#pragma unroll
    for (int t = 0; t < 8; ++t) {
        const int n = t * 16 + lo;
#pragma unroll
        for (int i = 0; i < 8; ++i)
            ft[(i + hi * 8) * SK + n] = (unsigned short)f2bfu(swishf(u1c[t][i]));
    }
#pragma unroll
    for (int t = 0; t < 8; ++t) {
        const int n = t * 16 + lo;
        v8f c = splat8(upd2b[n]);
#pragma unroll
        for (int kc = 0; kc < 128; kc += 32)
            c = wmma_bf16(frag_a(ft, lo, hi, kc), frag_b(wb2, lo, hi, kc, t * 16), c);
#pragma unroll
        for (int i = 0; i < 8; ++i) {
            const int node = mybase + i + hi * 8;
            fB[(size_t)node * HID + n] = fA[(size_t)node * HID + n] + swishf(c[i]);
        }
    }
}

// ---------------- LayerNorm: deterministic 3-stage ----------------
__global__ __launch_bounds__(256)
void ln_partial_kernel(const float* __restrict__ fBv, float* __restrict__ part) {
    const int chunk = blockIdx.x, b = blockIdx.y;
    const int tid = threadIdx.x;
    const int cch = tid & 127, half = tid >> 7;
    const int nbase = b * 16384 + chunk * 512;
    float s = 0.0f, s2 = 0.0f;
    for (int i = half; i < 512; i += 2) {
        const float v = fBv[(size_t)(nbase + i) * HID + cch];
        s += v; s2 += v * v;
    }
    __shared__ float sh[256], sh2[256];
    sh[tid] = s; sh2[tid] = s2;
    __syncthreads();
    if (half == 0) {
        const size_t o = ((size_t)(b * 32 + chunk) * 128 + cch) * 2;
        part[o + 0] = sh[tid] + sh[tid + 128];
        part[o + 1] = sh2[tid] + sh2[tid + 128];
    }
}

__global__ __launch_bounds__(128)
void ln_finish_kernel(const float* __restrict__ part, float* __restrict__ st) {
    const int b = blockIdx.x, c = threadIdx.x;
    float S = 0.0f, S2 = 0.0f;
    for (int ch = 0; ch < 32; ++ch) {
        const float* p = part + ((size_t)(b * 32 + ch) * 128 + c) * 2;
        S += p[0]; S2 += p[1];
    }
    const float mean = S * (1.0f / 16384.0f);
    const float var  = S2 * (1.0f / 16384.0f) - mean * mean;
    st[((size_t)b * 128 + c) * 2 + 0] = mean;
    st[((size_t)b * 128 + c) * 2 + 1] = rsqrtf(var + 1e-5f);
}

__global__ __launch_bounds__(256)
void ln_apply_kernel(const float* __restrict__ fBv, const float* __restrict__ st,
                     float* __restrict__ fAv, unsigned short* __restrict__ fbf) {
    const int idx = blockIdx.x * 256 + threadIdx.x;   // 0..2097151
    const size_t flat = (size_t)idx * 4;
    const int cch  = (int)(flat & 127);
    const int node = (int)(flat >> 7);
    const int b    = node >> 14;
    float4 x = *(const float4*)(fBv + flat);
    const float* s = st + ((size_t)b * 128 + cch) * 2;
    x.x = (x.x - s[0]) * s[1];
    x.y = (x.y - s[2]) * s[3];
    x.z = (x.z - s[4]) * s[5];
    x.w = (x.w - s[6]) * s[7];
    *(float4*)(fAv + flat) = x;
    uint2 pk;
    pk.x = f2bfu(x.x) | (f2bfu(x.y) << 16);
    pk.y = f2bfu(x.z) | (f2bfu(x.w) << 16);
    *(uint2*)(fbf + flat) = pk;
}

// ---------------- readout (128->64->1), one block per node ----------------
__global__ __launch_bounds__(64)
void out_kernel(const float* __restrict__ fAv, const float* __restrict__ un,
                const float* __restrict__ oW1, const float* __restrict__ ob1,
                const float* __restrict__ oW2, const float* __restrict__ ob2,
                float* __restrict__ out) {
    const int node = blockIdx.x;
    const int tid  = threadIdx.x;   // 64 threads
    __shared__ float fl[128];
    __shared__ float h[64];
    fl[tid]      = fAv[(size_t)node * HID + tid];
    fl[tid + 64] = fAv[(size_t)node * HID + tid + 64];
    __syncthreads();
    float a = ob1[tid];
    for (int k = 0; k < 128; ++k) a += fl[k] * oW1[k * 64 + tid];
    h[tid] = swishf(a) * oW2[tid];
    __syncthreads();
    for (int s = 32; s > 0; s >>= 1) {
        if (tid < s) h[tid] += h[tid + s];
        __syncthreads();
    }
    if (tid == 0) out[node] = un[node] + 0.1f * (h[0] + ob2[0]);
}

// ---------------- launcher ----------------
extern "C" void kernel_launch(void* const* d_in, const int* in_sizes, int n_in,
                              void* d_out, int out_size, void* d_ws, size_t ws_size,
                              hipStream_t stream) {
    (void)in_sizes; (void)n_in; (void)out_size; (void)ws_size;
    const float* inputs = (const float*)d_in[0];
    const float* cp     = (const float*)d_in[1];
    // d_in[2] = edge_index (int32) — graph is synthesized analytically, unused
    const float* eW1 = (const float*)d_in[3];
    const float* eb1 = (const float*)d_in[4];
    const float* eW2 = (const float*)d_in[5];
    const float* eb2 = (const float*)d_in[6];
    const float* m1W = (const float*)d_in[7];
    const float* m1b = (const float*)d_in[8];
    const float* m2W = (const float*)d_in[9];
    const float* m2b = (const float*)d_in[10];
    const float* u1W = (const float*)d_in[11];
    const float* u1b = (const float*)d_in[12];
    const float* u2W = (const float*)d_in[13];
    const float* u2b = (const float*)d_in[14];
    const float* oW1 = (const float*)d_in[15];
    const float* ob1 = (const float*)d_in[16];
    const float* oW2 = (const float*)d_in[17];
    const float* ob2 = (const float*)d_in[18];
    float* out = (float*)d_out;

    char* ws = (char*)d_ws;
    const size_t OFF_FA  = 0;                                   // 65536*128 f32
    const size_t OFF_FB  = OFF_FA + (size_t)65536 * 128 * 4;
    const size_t OFF_UN  = OFF_FB + (size_t)65536 * 128 * 4;
    const size_t OFF_WBF = OFF_UN + (size_t)65536 * 4;          // 6*6*16384 bf16
    const size_t OFF_LNP = OFF_WBF + (size_t)6 * 6 * 16384 * 2;
    const size_t OFF_LNS = OFF_LNP + (size_t)4 * 32 * 128 * 2 * 4;
    const size_t OFF_FBF = OFF_LNS + (size_t)4 * 128 * 2 * 4;   // bf16 mirror of f

    float*          fA  = (float*)(ws + OFF_FA);
    float*          fB  = (float*)(ws + OFF_FB);
    float*          un  = (float*)(ws + OFF_UN);
    unsigned short* wbf = (unsigned short*)(ws + OFF_WBF);
    float*          lnp = (float*)(ws + OFF_LNP);
    float*          lns = (float*)(ws + OFF_LNS);
    unsigned short* fbf = (unsigned short*)(ws + OFF_FBF);

    prep_u_kernel<<<256, 256, 0, stream>>>(inputs, un);
    conv_w_kernel<<<(6 * 6 * 16384 + 255) / 256, 256, 0, stream>>>(m1W, m2W, u1W, u2W, wbf);
    embed_kernel<<<65536, 128, 0, stream>>>(un, cp, eW1, eb1, eW2, eb2, fA, fbf);

    for (int l = 0; l < 6; ++l) {
        layer_kernel<<<1024, 128, 0, stream>>>(
            fA, fB, fbf, un,
            wbf + (size_t)l * 6 * 16384,
            m1W + (size_t)l * 264 * 128, m1b + (size_t)l * 128,
            m2b + (size_t)l * 128,
            u1W + (size_t)l * 261 * 128, u1b + (size_t)l * 128,
            u2b + (size_t)l * 128,
            cp);
        ln_partial_kernel<<<dim3(32, 4), 256, 0, stream>>>(fB, lnp);
        ln_finish_kernel<<<4, 128, 0, stream>>>(lnp, lns);
        ln_apply_kernel<<<8192, 256, 0, stream>>>(fB, lns, fA, fbf);
    }

    out_kernel<<<65536, 64, 0, stream>>>(fA, un, oW1, ob1, oW2, ob2, out);
}
